// GraphIpaDenoisingLayer_66159676228220
// MI455X (gfx1250) — matrix operand order repaired and checked by reading, not compiled, not run on hardware
//
#include <hip/hip_runtime.h>
#include <hip/hip_bf16.h>
#include <cstring>

typedef __attribute__((ext_vector_type(16))) _Float16 v16h;
typedef __attribute__((ext_vector_type(8)))  _Float16 v8h;
typedef __attribute__((ext_vector_type(8)))  float    v8f;
typedef __attribute__((ext_vector_type(4)))  unsigned v4u;
typedef __attribute__((ext_vector_type(8)))  unsigned v8u;

#define NND  8192
#define EED  131072
#define BBD  32
#define HHD  12
#define CHDD 16
#define PQD  4
#define PVD  8
#define CSD  384
#define CZD  64
#define CLATD 128
#define HIDD 448
#define DHDD 32
#define CATD 1344   // H*(CH + PV*4 + CZ) = 12*112
#define EB   128    // edges per block in fused edge MLP
#define LDSZ_OFF (2 * EB * HIDD * 2)   // byte offset of z staging region = 229376

#define SHUF16(a0, a1) __builtin_shufflevector(a0, a1, 0,1,2,3,4,5,6,7,8,9,10,11,12,13,14,15)
#define WMMA16(a, b, c) __builtin_amdgcn_wmma_f32_16x16x32_f16(false, a, false, b, (short)0, c, false, false)

// ---------------- helpers ----------------
__device__ __forceinline__ unsigned ordEnc(float f) {
    unsigned u = __float_as_uint(f);
    return (u & 0x80000000u) ? ~u : (u | 0x80000000u);
}
__device__ __forceinline__ float ordDec(unsigned u) {
    unsigned b = (u & 0x80000000u) ? (u & 0x7fffffffu) : ~u;
    return __uint_as_float(b);
}

// ---------------- tiny elementwise kernels ----------------
__global__ void fill_u32_kernel(unsigned* p, unsigned v, long n) {
    long t = (long)blockIdx.x * blockDim.x + threadIdx.x;
    if (t < n) p[t] = v;
}
__global__ void cast_f16_kernel(const float* s, _Float16* d, long n) {
    long t = (long)blockIdx.x * blockDim.x + threadIdx.x;
    if (t < n) d[t] = (_Float16)s[t];
}
// src [K,N] f32 row-major -> dst [N,K] f16 (transposed)
__global__ void cast_tr_kernel(const float* src, _Float16* dst, int K, int N) {
    long t = (long)blockIdx.x * blockDim.x + threadIdx.x;
    if (t >= (long)K * N) return;
    int n = (int)(t / K), k = (int)(t - (long)n * K);
    dst[t] = (_Float16)src[(long)k * N + n];
}
// ipa_wb [64,12] -> padded transposed [16,64]
__global__ void wb_padtr_kernel(const float* src, _Float16* dst) {
    int t = blockIdx.x * blockDim.x + threadIdx.x;
    if (t >= 16 * 64) return;
    int n = t / 64, k = t - n * 64;
    dst[t] = (n < 12) ? (_Float16)src[k * 12 + n] : (_Float16)0.0f;
}
__global__ void maskf_kernel(const int* m, float* o, int n) {
    int t = blockIdx.x * blockDim.x + threadIdx.x;
    if (t < n) o[t] = (float)m[t];
}
__global__ void concat_fuse_kernel(const float* nf, const float* sc, _Float16* a, int n) {
    long t = (long)blockIdx.x * blockDim.x + threadIdx.x;
    if (t >= (long)n * 512) return;
    int row = (int)(t / 512), c = (int)(t - (long)row * 512);
    float v = (c < CSD) ? nf[(long)row * CSD + c] : sc[(long)row * CLATD + (c - CSD)];
    a[t] = (_Float16)v;
}
__global__ void add_rowmask_kernel(const float* a, const float* b, const float* mask,
                                   float* o, int rows, int C) {
    long t = (long)blockIdx.x * blockDim.x + threadIdx.x;
    if (t >= (long)rows * C) return;
    int row = (int)(t / C);
    o[t] = a[t] + b[t] * mask[row];
}
__global__ void add_kernel(const float* a, const float* b, float* o, long n) {
    long t = (long)blockIdx.x * blockDim.x + threadIdx.x;
    if (t < n) o[t] = a[t] + b[t];
}
__global__ void gather_add_kernel(const float* node1, const float* vnb, const int* batch,
                                  const float* mask, float* o, _Float16* o16, int rows, int C) {
    long t = (long)blockIdx.x * blockDim.x + threadIdx.x;
    if (t >= (long)rows * C) return;
    int row = (int)(t / C), c = (int)(t - (long)row * C);
    float v = node1[t] + vnb[(long)batch[row] * C + c] * mask[row];
    o[t] = v; o16[t] = (_Float16)v;
}
__global__ void vn_add_kernel(const float* vn0, const float* d, float* o, _Float16* o16, long n) {
    long t = (long)blockIdx.x * blockDim.x + threadIdx.x;
    if (t < n) { float v = vn0[t] + d[t]; o[t] = v; o16[t] = (_Float16)v; }
}
__global__ void make_nm16_kernel(const float* node, const float* nm, _Float16* o, int rows, int C) {
    long t = (long)blockIdx.x * blockDim.x + threadIdx.x;
    if (t >= (long)rows * C) return;
    int row = (int)(t / C);
    o[t] = (_Float16)(node[t] * nm[row]);
}
__global__ void sc_epilogue_kernel(const float* sc, const float* scout, const float* b,
                                   const float* nm, float* o, int rows, int C) {
    long t = (long)blockIdx.x * blockDim.x + threadIdx.x;
    if (t >= (long)rows * C) return;
    int row = (int)(t / C), c = (int)(t - (long)row * C);
    o[t] = sc[t] + (scout[t] + b[c]) * nm[row];
}
__global__ void to_global_kernel(float* raw, const float* rot, const float* trans, int HP, int n) {
    int t = blockIdx.x * blockDim.x + threadIdx.x;
    if (t >= n * HP) return;
    int nn = t / HP, hp = t - nn * HP;
    const float* R = rot + (long)nn * 9;
    const float* T = trans + (long)nn * 3;
    float* p = raw + (long)nn * HP * 3 + (long)hp * 3;
    float y0 = p[0], y1 = p[1], y2 = p[2];
    p[0] = R[0]*y0 + R[1]*y1 + R[2]*y2 + T[0];
    p[1] = R[3]*y0 + R[4]*y1 + R[5]*y2 + T[1];
    p[2] = R[6]*y0 + R[7]*y1 + R[8]*y2 + T[2];
}

// ---------------- layernorm (wave per row, wave32 shuffles) ----------------
__global__ void ln_kernel(const float* x, const float* g, const float* b,
                          const float* rowScale, float* outF, _Float16* outH,
                          int rows, int C) {
    int wpb = blockDim.x >> 5;
    int row = blockIdx.x * wpb + (threadIdx.x >> 5);
    if (row >= rows) return;
    int lane = threadIdx.x & 31;
    const float* xr = x + (long)row * C;
    float s = 0.f, s2 = 0.f;
    for (int c = lane; c < C; c += 32) { float v = xr[c]; s += v; s2 += v * v; }
    #pragma unroll
    for (int m = 16; m; m >>= 1) { s += __shfl_xor(s, m, 32); s2 += __shfl_xor(s2, m, 32); }
    float mean = s / C;
    float var  = s2 / C - mean * mean;
    float inv  = rsqrtf(var + 1e-5f);
    float rs   = rowScale ? rowScale[row] : 1.0f;
    for (int c = lane; c < C; c += 32) {
        float v = ((xr[c] - mean) * inv * g[c] + b[c]) * rs;
        long oi = (long)row * C + c;
        if (outF) outF[oi] = v;
        if (outH) outH[oi] = (_Float16)v;
    }
}

// ---------------- generic WMMA GEMM ----------------
// A [M,K] f16 row-major; BT [N,K] f16 (transposed weights); M%16==0, N%16==0, K%32==0.
__global__ __launch_bounds__(256)
void wmma_gemm_kernel(const _Float16* __restrict__ A, const _Float16* __restrict__ BT,
                      const float* __restrict__ bias, float* outF, _Float16* outH,
                      int M, int N, int K, int relu) {
    int wave = threadIdx.x >> 5, lane = threadIdx.x & 31;
    int NT = N >> 4, MT = M >> 4;
    int tileId = blockIdx.x * 8 + wave;
    if (tileId >= MT * NT) return;
    int mt = tileId / NT, nt = tileId - mt * NT;
    int idx = lane & 15, half = lane >> 4;
    int col = nt * 16 + idx;
    const _Float16* Arow = A + (long)(mt * 16 + idx) * K + (half << 4);
    const _Float16* Bcol = BT + (long)col * K + (half << 4);
    v8f acc = {};
    for (int kk = 0; kk < K; kk += 32) {
        v16h a = *(const v16h*)(Arow + kk);         // A: K = kk+16*half .. +15
        v16h b = *(const v16h*)(Bcol + kk);         // B: K = kk+16*half .. +15
        acc = WMMA16(a, b, acc);
    }
    float bv = bias ? bias[col] : 0.0f;
    #pragma unroll
    for (int r = 0; r < 8; ++r) {
        int orow = mt * 16 + r + 8 * half;
        float v = acc[r] + bv;
        if (relu) v = fmaxf(v, 0.0f);
        long oi = (long)orow * N + col;
        if (outF) outF[oi] = v;
        if (outH) outH[oi] = (_Float16)v;
    }
}

// ---------------- IPA attention kernels ----------------
__global__ void att_kernel(const float* Q, const float* Kf, const float* qp, const float* kp,
                           const float* zb, const float* hwp, const float* maskf,
                           const int* ei, float* att, unsigned* segmax) {
    int e = blockIdx.x * blockDim.x + threadIdx.x;
    if (e >= EED) return;
    int i = ei[e], j = ei[EED + e];
    float mterm = (maskf[j] - 1.0f) * 1e9f;
    for (int h = 0; h < HHD; ++h) {
        float dot = 0.f;
        #pragma unroll
        for (int c = 0; c < CHDD; ++c)
            dot += Q[(long)i * 192 + h * 16 + c] * Kf[(long)j * 192 + h * 16 + c];
        float a = dot * 0.14433757f;                 // sqrt(1/48)
        a += 0.57735027f * zb[(long)e * 16 + h];     // sqrt(1/3); zb padded to stride 16
        float d2s = 0.f;
        #pragma unroll
        for (int pc = 0; pc < PQD * 3; ++pc) {
            float d = qp[(long)i * 144 + h * 12 + pc] - kp[(long)j * 144 + h * 12 + pc];
            d2s += d * d;
        }
        float hw = log1pf(expf(hwp[h])) * 0.13608276f;   // softplus * sqrt(1/54)
        a += d2s * hw * -0.5f;
        a += mterm;
        att[(long)e * HHD + h] = a;
        atomicMax(&segmax[(long)i * HHD + h], ordEnc(a));
    }
}
__global__ void exp_kernel(const int* ei, float* att, const unsigned* segmax, float* den) {
    long t = (long)blockIdx.x * blockDim.x + threadIdx.x;
    if (t >= (long)EED * HHD) return;
    int e = (int)(t / HHD), h = (int)(t - (long)e * HHD);
    int i = ei[e];
    float ex = expf(att[t] - ordDec(segmax[(long)i * HHD + h]));
    att[t] = ex;
    atomicAdd(&den[(long)i * HHD + h], ex);
}
__global__ void accum_kernel(const int* ei, const float* att, const float* den,
                             const float* V, const float* vp, const float* zf,
                             float* o_acc, float* optg, float* opair) {
    long t = (long)blockIdx.x * blockDim.x + threadIdx.x;
    if (t >= (long)EED * HHD) return;
    int e = (int)(t / HHD), h = (int)(t - (long)e * HHD);
    int i = ei[e], j = ei[EED + e];
    float a = att[t] / (den[(long)i * HHD + h] + 1e-9f);
    #pragma unroll
    for (int c = 0; c < CHDD; ++c)
        atomicAdd(&o_acc[(long)i * 192 + h * 16 + c], a * V[(long)j * 192 + h * 16 + c]);
    #pragma unroll
    for (int pc = 0; pc < PVD * 3; ++pc)
        atomicAdd(&optg[(long)i * 288 + h * 24 + pc], a * vp[(long)j * 288 + h * 24 + pc]);
    for (int zc = 0; zc < CZD; ++zc)
        atomicAdd(&opair[(long)i * 768 + h * 64 + zc], a * zf[(long)e * CZD + zc]);
}
__global__ void finalize_cat_kernel(const float* o_acc, const float* optg, const float* opair,
                                    const float* rot, const float* trans, _Float16* cat16) {
    int t = blockIdx.x * blockDim.x + threadIdx.x;
    if (t >= NND * HHD) return;
    int n = t / HHD, h = t - n * HHD;
    const float* R = rot + (long)n * 9;
    const float* T = trans + (long)n * 3;
    _Float16* cr = cat16 + (long)n * CATD;
    #pragma unroll
    for (int c = 0; c < CHDD; ++c)
        cr[h * 16 + c] = (_Float16)o_acc[(long)n * 192 + h * 16 + c];
    #pragma unroll
    for (int p = 0; p < PVD; ++p) {
        float g0 = optg[(long)n * 288 + (h * PVD + p) * 3 + 0] - T[0];
        float g1 = optg[(long)n * 288 + (h * PVD + p) * 3 + 1] - T[1];
        float g2 = optg[(long)n * 288 + (h * PVD + p) * 3 + 2] - T[2];
        float o0 = R[0] * g0 + R[3] * g1 + R[6] * g2;    // rot^T (einsum 'nyx')
        float o1 = R[1] * g0 + R[4] * g1 + R[7] * g2;
        float o2 = R[2] * g0 + R[5] * g1 + R[8] * g2;
        cr[192 + (h * PVD + p) * 3 + 0] = (_Float16)o0;
        cr[192 + (h * PVD + p) * 3 + 1] = (_Float16)o1;
        cr[192 + (h * PVD + p) * 3 + 2] = (_Float16)o2;
        cr[480 + h * PVD + p] = (_Float16)sqrtf(o0*o0 + o1*o1 + o2*o2 + 1e-8f);
    }
    for (int zc = 0; zc < CZD; ++zc)
        cr[576 + h * 64 + zc] = (_Float16)opair[(long)n * 768 + h * 64 + zc];
}

// ---------------- virtual-node attention ----------------
__global__ void vn_logits_kernel(const float* q2, const float* k2, const int* batch,
                                 const float* maskf, float* lg, unsigned* bmax) {
    int t = blockIdx.x * blockDim.x + threadIdx.x;
    if (t >= NND * HHD) return;
    int n = t / HHD, h = t - n * HHD;
    int b = batch[n];
    float dot = 0.f;
    #pragma unroll
    for (int d = 0; d < DHDD; ++d)
        dot += q2[(long)b * CSD + h * DHDD + d] * k2[(long)n * CSD + h * DHDD + d];
    float l = dot * 0.17677670f + (maskf[n] - 1.0f) * 1e9f;
    lg[t] = l;
    atomicMax(&bmax[b * HHD + h], ordEnc(l));
}
__global__ void vn_exp_kernel(const int* batch, float* lg, const unsigned* bmax, float* bden) {
    int t = blockIdx.x * blockDim.x + threadIdx.x;
    if (t >= NND * HHD) return;
    int n = t / HHD, h = t - n * HHD;
    int b = batch[n];
    float ex = expf(lg[t] - ordDec(bmax[b * HHD + h]));
    lg[t] = ex;
    atomicAdd(&bden[b * HHD + h], ex);
}
__global__ void vn_accum_kernel(const int* batch, const float* lg, const float* bden,
                                const float* v2, float* vnout) {
    int t = blockIdx.x * blockDim.x + threadIdx.x;
    if (t >= NND * HHD) return;
    int n = t / HHD, h = t - n * HHD;
    int b = batch[n];
    float a = lg[t] / (bden[b * HHD + h] + 1e-9f);
    #pragma unroll
    for (int d = 0; d < DHDD; ++d)
        atomicAdd(&vnout[(long)b * CSD + h * DHDD + d], a * v2[(long)n * CSD + h * DHDD + d]);
}

// ---------------- backbone head + quaternion compose ----------------
__global__ void bb_compose_kernel(const float* node, const float* nm,
                                  const float* bbw, const float* bbb,
                                  const float* rot, const float* trans,
                                  float* rot2, float* trans2, int n) {
    int t = blockIdx.x * blockDim.x + threadIdx.x;
    if (t >= n) return;
    float nmv = nm[t];
    float uv[6];
    #pragma unroll
    for (int c = 0; c < 6; ++c) uv[c] = 0.f;
    const float* nr = node + (long)t * CSD;
    for (int k = 0; k < CSD; ++k) {
        float xv = nr[k] * nmv;
        #pragma unroll
        for (int c = 0; c < 6; ++c) uv[c] += xv * bbw[k * 6 + c];
    }
    #pragma unroll
    for (int c = 0; c < 6; ++c) uv[c] = (uv[c] + bbb[c]) * nmv;
    float qb = uv[0], qc = uv[1], qd = uv[2];
    float inr = rsqrtf(1.f + qb*qb + qc*qc + qd*qd);
    float w = inr, x = qb*inr, y = qc*inr, z = qd*inr;
    float Rm[9];
    Rm[0]=1-2*(y*y+z*z); Rm[1]=2*(x*y-w*z); Rm[2]=2*(x*z+w*y);
    Rm[3]=2*(x*y+w*z);   Rm[4]=1-2*(x*x+z*z); Rm[5]=2*(y*z-w*x);
    Rm[6]=2*(x*z-w*y);   Rm[7]=2*(y*z+w*x);   Rm[8]=1-2*(x*x+y*y);
    const float* R0 = rot + (long)t * 9;
    float* Ro = rot2 + (long)t * 9;
    #pragma unroll
    for (int i2 = 0; i2 < 3; ++i2)
        #pragma unroll
        for (int k2 = 0; k2 < 3; ++k2)
            Ro[i2*3+k2] = R0[i2*3+0]*Rm[0*3+k2] + R0[i2*3+1]*Rm[1*3+k2] + R0[i2*3+2]*Rm[2*3+k2];
    const float* T0 = trans + (long)t * 3;
    float* To = trans2 + (long)t * 3;
    #pragma unroll
    for (int i2 = 0; i2 < 3; ++i2)
        To[i2] = T0[i2] + R0[i2*3+0]*uv[3] + R0[i2*3+1]*uv[4] + R0[i2*3+2]*uv[5];
}

// ---------------- fused edge-transition MLP (LDS-resident, WMMA, 2x2 quads) ----------------
// smIn [EB,448] f16, WT [448,448] f16 transposed. 8 M-tiles x 28 N-tiles as 4x14 quads.
__device__ __forceinline__ void edge_stage(const _Float16* smIn, const _Float16* __restrict__ WT,
                                           const float* __restrict__ bias, _Float16* smOut,
                                           int wave, int lane) {
    int idx = lane & 15, half = lane >> 4;
    for (int t = wave; t < 56; t += 8) {
        int mp = t & 3, np = t >> 2;
        int row0 = (mp * 2) * 16 + idx;
        int col0 = np * 32 + idx, col1 = col0 + 16;
        v8f acc00 = {}, acc01 = {}, acc10 = {}, acc11 = {};
        const _Float16* Ar0 = smIn + row0 * HIDD + (half << 4);
        const _Float16* Ar1 = Ar0 + 16 * HIDD;
        const _Float16* Bc0 = WT + (long)col0 * HIDD + (half << 4);
        const _Float16* Bc1 = WT + (long)col1 * HIDD + (half << 4);
        for (int kk = 0; kk < HIDD; kk += 32) {
            v16h a0 = *(const v16h*)(Ar0 + kk);
            v16h a1 = *(const v16h*)(Ar1 + kk);
            v16h b0 = *(const v16h*)(Bc0 + kk);
            v16h b1 = *(const v16h*)(Bc1 + kk);
            acc00 = WMMA16(a0, b0, acc00);
            acc01 = WMMA16(a0, b1, acc01);
            acc10 = WMMA16(a1, b0, acc10);
            acc11 = WMMA16(a1, b1, acc11);
        }
        float bv0 = bias[col0], bv1 = bias[col1];
        #pragma unroll
        for (int r = 0; r < 8; ++r) {
            int or0 = row0 - idx + r + 8 * half;
            int or1 = or0 + 16;
            smOut[or0 * HIDD + col0] = (_Float16)fmaxf(acc00[r] + bv0, 0.0f);
            smOut[or0 * HIDD + col1] = (_Float16)fmaxf(acc01[r] + bv1, 0.0f);
            smOut[or1 * HIDD + col0] = (_Float16)fmaxf(acc10[r] + bv0, 0.0f);
            smOut[or1 * HIDD + col1] = (_Float16)fmaxf(acc11[r] + bv1, 0.0f);
        }
    }
}

__global__ __launch_bounds__(256)
void edge_mlp_kernel(const _Float16* __restrict__ zq, const _Float16* __restrict__ nb16,
                     const int* __restrict__ ei,
                     const _Float16* __restrict__ W1T, const float* __restrict__ b1,
                     const _Float16* __restrict__ W2T, const float* __restrict__ b2,
                     const _Float16* __restrict__ W3T, const float* __restrict__ b3,
                     const float* __restrict__ g, const float* __restrict__ bln,
                     float* __restrict__ out) {
    extern __shared__ char smem[];
    _Float16* smA = (_Float16*)smem;                 // EB x 448 f16
    _Float16* smB = smA + EB * HIDD;                 // EB x 448 f16
    _Float16* smZ = smB + EB * HIDD;                 // EB x 64  f16 (TDM staging)
    float*    smF = (float*)smB;                     // EB x 64  f32 (reuses smB)
    int tid = threadIdx.x, lane = tid & 31, wave = tid >> 5;
    int e0 = blockIdx.x * EB;

    // ---- TDM: async DMA the contiguous EBx64 f16 z tile into LDS staging ----
    if (wave == 0) {
        unsigned long long ga = (unsigned long long)(const void*)(zq + (size_t)e0 * CZD);
        v4u g0; v8u g1;
        g0[0] = 1u;                                        // count=1 (valid), user mode
        g0[1] = (unsigned)LDSZ_OFF;                        // lds_addr (bytes)
        g0[2] = (unsigned)(ga & 0xffffffffu);              // global_addr[31:0]
        g0[3] = (unsigned)((ga >> 32) & 0x01ffffffu) | 0x80000000u; // addr[56:32] | type=2
        g1[0] = 0x00010000u;                               // data_size=1 -> 2 bytes/elem
        g1[1] = (unsigned)CZD << 16;                       // tensor_dim0 = 64 (bits 63:48)
        g1[2] = 0u;                                        // dim0 hi / dim1 lo
        g1[3] = 2u | ((unsigned)CZD << 16);                // tensor_dim1 = 131072; tile_dim0 = 64
        g1[4] = (unsigned)EB;                              // tile_dim1 = 128
        g1[5] = (unsigned)CZD;                             // tensor_dim0_stride = 64
        g1[6] = 0u;
        g1[7] = 0u;
        asm volatile("tensor_load_to_lds %0, %1" :: "s"(g0), "s"(g1) : "memory");
    }
    // L2 prefetch of the streamed weight matrices while DMA is in flight
    for (int t = tid; t < (HIDD * HIDD) / 64; t += 256) {
        __builtin_prefetch(W1T + t * 64, 0, 1);
        __builtin_prefetch(W2T + t * 64, 0, 1);
    }
    // gather nb[i] / nb[j] columns of h0 while the TDM runs
    for (int t = tid; t < EB * HIDD; t += 256) {
        int r = t / HIDD, c = t - r * HIDD;
        if (c < CZD) continue;
        int e = e0 + r;
        smA[t] = (c < 256) ? nb16[(long)ei[e] * 192 + (c - 64)]
                           : nb16[(long)ei[EED + e] * 192 + (c - 256)];
    }
    if (wave == 0) __builtin_amdgcn_s_wait_tensorcnt(0);
    __syncthreads();
    // scatter staged z into the h0 row layout
    for (int t = tid; t < EB * CZD; t += 256) {
        int r = t >> 6, c = t & 63;
        smA[r * HIDD + c] = smZ[t];
    }
    __syncthreads();
    edge_stage(smA, W1T, b1, smB, wave, lane);       // h1 = relu(h0 @ W1 + b1)
    __syncthreads();
    edge_stage(smB, W2T, b2, smA, wave, lane);       // h2 = relu(h1 @ W2 + b2)
    __syncthreads();
    // h3 = h2 @ W3 + b3 -> smF f32; 8 units (4 mtPairs x 2 ntPairs), one quad per wave
    {
        int idx = lane & 15, half = lane >> 4;
        int mp = wave & 3, np = wave >> 2;
        int row0 = (mp * 2) * 16 + idx;
        int col0 = np * 32 + idx, col1 = col0 + 16;
        v8f acc00 = {}, acc01 = {}, acc10 = {}, acc11 = {};
        const _Float16* Ar0 = smA + row0 * HIDD + (half << 4);
        const _Float16* Ar1 = Ar0 + 16 * HIDD;
        const _Float16* Bc0 = W3T + (long)col0 * HIDD + (half << 4);
        const _Float16* Bc1 = W3T + (long)col1 * HIDD + (half << 4);
        for (int kk = 0; kk < HIDD; kk += 32) {
            v16h a0 = *(const v16h*)(Ar0 + kk);
            v16h a1 = *(const v16h*)(Ar1 + kk);
            v16h b0 = *(const v16h*)(Bc0 + kk);
            v16h b1 = *(const v16h*)(Bc1 + kk);
            acc00 = WMMA16(a0, b0, acc00);
            acc01 = WMMA16(a0, b1, acc01);
            acc10 = WMMA16(a1, b0, acc10);
            acc11 = WMMA16(a1, b1, acc11);
        }
        float bv0 = b3[col0], bv1 = b3[col1];
        #pragma unroll
        for (int r = 0; r < 8; ++r) {
            int or0 = (mp * 2) * 16 + r + 8 * half;
            int or1 = or0 + 16;
            smF[or0 * CZD + col0] = acc00[r] + bv0;
            smF[or0 * CZD + col1] = acc01[r] + bv1;
            smF[or1 * CZD + col0] = acc10[r] + bv0;
            smF[or1 * CZD + col1] = acc11[r] + bv1;
        }
    }
    __syncthreads();
    // layernorm over 64, 16 rows per wave, write to global
    for (int rr = 0; rr < 16; ++rr) {
        int row = wave * 16 + rr;
        float v0 = smF[row * 64 + lane], v1 = smF[row * 64 + 32 + lane];
        float s = v0 + v1, s2 = v0 * v0 + v1 * v1;
        #pragma unroll
        for (int m = 16; m; m >>= 1) { s += __shfl_xor(s, m, 32); s2 += __shfl_xor(s2, m, 32); }
        float mean = s * (1.f / 64.f);
        float var  = s2 * (1.f / 64.f) - mean * mean;
        float inv  = rsqrtf(var + 1e-5f);
        long o = (long)(e0 + row) * 64;
        out[o + lane]      = (v0 - mean) * inv * g[lane]      + bln[lane];
        out[o + 32 + lane] = (v1 - mean) * inv * g[32 + lane] + bln[32 + lane];
    }
}

// ---------------- host launcher ----------------
extern "C" void kernel_launch(void* const* d_in, const int* in_sizes, int n_in,
                              void* d_out, int out_size, void* d_ws, size_t ws_size,
                              hipStream_t stream) {
    (void)in_sizes; (void)n_in; (void)out_size; (void)ws_size;
    const float* node_features = (const float*)d_in[0];
    const float* vn_features   = (const float*)d_in[1];
    const float* rot           = (const float*)d_in[2];
    const float* trans         = (const float*)d_in[3];
    const float* sidechain     = (const float*)d_in[4];
    const float* edge_feat     = (const float*)d_in[5];
    const int*   edge_index    = (const int*)d_in[6];
    const int*   res_mask      = (const int*)d_in[7];
    const float* noising       = (const float*)d_in[8];
    const int*   batch         = (const int*)d_in[9];
    const float* fuse_w = (const float*)d_in[10]; const float* fuse_b = (const float*)d_in[11];
    const float* ipa_wq = (const float*)d_in[12]; const float* ipa_bq = (const float*)d_in[13];
    const float* ipa_wk = (const float*)d_in[14]; const float* ipa_bk = (const float*)d_in[15];
    const float* ipa_wv = (const float*)d_in[16]; const float* ipa_bv = (const float*)d_in[17];
    const float* ipa_wqp = (const float*)d_in[18];
    const float* ipa_wkp = (const float*)d_in[19];
    const float* ipa_wvp = (const float*)d_in[20];
    const float* ipa_wb  = (const float*)d_in[21];
    const float* ipa_hw  = (const float*)d_in[22];
    const float* ipa_wo  = (const float*)d_in[23]; const float* ipa_bo = (const float*)d_in[24];
    const float* ln1_g = (const float*)d_in[25]; const float* ln1_b = (const float*)d_in[26];
    const float* vn_wq = (const float*)d_in[27]; const float* vn_wk = (const float*)d_in[28];
    const float* vn_wv = (const float*)d_in[29]; const float* vn_wo = (const float*)d_in[30];
    const float* vn_wb = (const float*)d_in[31];
    const float* nt_w1 = (const float*)d_in[32]; const float* nt_b1 = (const float*)d_in[33];
    const float* nt_w2 = (const float*)d_in[34]; const float* nt_b2 = (const float*)d_in[35];
    const float* nt_w3 = (const float*)d_in[36]; const float* nt_b3 = (const float*)d_in[37];
    const float* nt_g = (const float*)d_in[38];  const float* nt_bln = (const float*)d_in[39];
    const float* bb_w = (const float*)d_in[40];  const float* bb_b = (const float*)d_in[41];
    const float* sc_w = (const float*)d_in[42];  const float* sc_b = (const float*)d_in[43];
    const float* et_wn = (const float*)d_in[44]; const float* et_bn2 = (const float*)d_in[45];
    const float* et_w1 = (const float*)d_in[46]; const float* et_b1 = (const float*)d_in[47];
    const float* et_w2 = (const float*)d_in[48]; const float* et_b2 = (const float*)d_in[49];
    const float* et_w3 = (const float*)d_in[50]; const float* et_b3 = (const float*)d_in[51];
    const float* et_g  = (const float*)d_in[52]; const float* et_bln = (const float*)d_in[53];

    float* out  = (float*)d_out;
    float* o_node = out;
    float* o_vn   = out + (long)NND * CSD;
    float* o_rot  = o_vn + (long)BBD * CSD;
    float* o_tr   = o_rot + (long)NND * 9;
    float* o_sc   = o_tr + (long)NND * 3;
    float* o_edge = o_sc + (long)NND * CLATD;

    char* wsb = (char*)d_ws; size_t off = 0;
    auto alloc = [&](size_t bytes) -> void* {
        void* p = wsb + off; off = (off + bytes + 255) & ~(size_t)255; return p;
    };
    auto H16 = [&](size_t n) { return (_Float16*)alloc(n * 2); };
    auto F32 = [&](size_t n) { return (float*)alloc(n * 4); };
    auto U32 = [&](size_t n) { return (unsigned*)alloc(n * 4); };

    _Float16* fuse_wT = H16(512 * 384);
    _Float16* wqT = H16(384 * 192); _Float16* wkT = H16(384 * 192); _Float16* wvT = H16(384 * 192);
    _Float16* wqpT = H16(384 * 144); _Float16* wkpT = H16(384 * 144); _Float16* wvpT = H16(384 * 288);
    _Float16* wbT = H16(16 * 64);
    _Float16* woT = H16((size_t)CATD * 384);
    _Float16* vnqT = H16(384 * 384); _Float16* vnkT = H16(384 * 384); _Float16* vnvT = H16(384 * 384);
    _Float16* vnwoT = H16(384 * 384); _Float16* vnwbT = H16(384 * 384);
    _Float16* ntw1T = H16(384 * 384); _Float16* ntw2T = H16(384 * 384); _Float16* ntw3T = H16(384 * 384);
    _Float16* scwT = H16(384 * 128);
    _Float16* etwnT = H16(384 * 192);
    _Float16* etw1T = H16(448 * 448); _Float16* etw2T = H16(448 * 448); _Float16* etw3T = H16(448 * 64);
    float* maskf = F32(NND);
    _Float16* Afuse16 = H16((size_t)NND * 512);
    _Float16* s16 = H16((size_t)NND * 384);
    float* Qf = F32((size_t)NND * 192); float* Kf = F32((size_t)NND * 192); float* Vf = F32((size_t)NND * 192);
    float* qp = F32((size_t)NND * 144); float* kp = F32((size_t)NND * 144); float* vp = F32((size_t)NND * 288);
    _Float16* zE16 = H16((size_t)EED * 64);
    float* zb = F32((size_t)EED * 16);
    float* att = F32((size_t)EED * 12);
    unsigned* segmax = U32((size_t)NND * 12);
    float* segden = F32((size_t)NND * 12);
    float* o_acc = F32((size_t)NND * 192);
    float* optg  = F32((size_t)NND * 288);
    float* opair = F32((size_t)NND * 768);
    _Float16* cat16 = H16((size_t)NND * CATD);
    float* upd = F32((size_t)NND * 384);
    float* pre1 = F32((size_t)NND * 384);
    float* node1 = F32((size_t)NND * 384); _Float16* node1_16 = H16((size_t)NND * 384);
    _Float16* vn16 = H16((size_t)BBD * 384);
    float* q2 = F32((size_t)BBD * 384);
    float* k2 = F32((size_t)NND * 384); float* v2 = F32((size_t)NND * 384);
    float* lg2 = F32((size_t)NND * 12);
    unsigned* bmax = U32(BBD * 12); float* bden = F32(BBD * 12);
    float* vnout = F32((size_t)BBD * 384); _Float16* vnout16 = H16((size_t)BBD * 384);
    float* vno_o = F32((size_t)BBD * 384); _Float16* vnNew16 = H16((size_t)BBD * 384);
    float* vnb = F32((size_t)BBD * 384);
    float* node2 = F32((size_t)NND * 384); _Float16* node2_16 = H16((size_t)NND * 384);
    _Float16* hA16 = H16((size_t)NND * 384); _Float16* hB16 = H16((size_t)NND * 384);
    float* ht = F32((size_t)NND * 384); float* pre2 = F32((size_t)NND * 384);
    _Float16* nodeF16 = H16((size_t)NND * 384); _Float16* nodeNM16 = H16((size_t)NND * 384);
    float* scout = F32((size_t)NND * 128);
    _Float16* nb16 = H16((size_t)NND * 192);

    unsigned encNeg;
    { float f = -3.0e38f; unsigned u; std::memcpy(&u, &f, 4);
      encNeg = (u & 0x80000000u) ? ~u : (u | 0x80000000u); }

    auto cast16 = [&](const float* s, _Float16* d, long n) {
        cast_f16_kernel<<<dim3((unsigned)((n + 255) / 256)), dim3(256), 0, stream>>>(s, d, n);
    };
    auto castTr = [&](const float* s, _Float16* d, int K, int N) {
        long n = (long)K * N;
        cast_tr_kernel<<<dim3((unsigned)((n + 255) / 256)), dim3(256), 0, stream>>>(s, d, K, N);
    };
    auto fillu = [&](unsigned* p, unsigned v, long n) {
        fill_u32_kernel<<<dim3((unsigned)((n + 255) / 256)), dim3(256), 0, stream>>>(p, v, n);
    };
    auto gemm = [&](const _Float16* A, const _Float16* BT, const float* bias,
                    float* oF, _Float16* oH, int M, int Nc, int Kc, int relu) {
        int tiles = (M / 16) * (Nc / 16);
        wmma_gemm_kernel<<<dim3((tiles + 7) / 8), dim3(256), 0, stream>>>(
            A, BT, bias, oF, oH, M, Nc, Kc, relu);
    };

    // ---- weight casts (transposed) ----
    castTr(fuse_w, fuse_wT, 512, 384);
    castTr(ipa_wq, wqT, 384, 192); castTr(ipa_wk, wkT, 384, 192); castTr(ipa_wv, wvT, 384, 192);
    castTr(ipa_wqp, wqpT, 384, 144); castTr(ipa_wkp, wkpT, 384, 144); castTr(ipa_wvp, wvpT, 384, 288);
    wb_padtr_kernel<<<dim3(4), dim3(256), 0, stream>>>(ipa_wb, wbT);
    castTr(ipa_wo, woT, CATD, 384);
    castTr(vn_wq, vnqT, 384, 384); castTr(vn_wk, vnkT, 384, 384); castTr(vn_wv, vnvT, 384, 384);
    castTr(vn_wo, vnwoT, 384, 384); castTr(vn_wb, vnwbT, 384, 384);
    castTr(nt_w1, ntw1T, 384, 384); castTr(nt_w2, ntw2T, 384, 384); castTr(nt_w3, ntw3T, 384, 384);
    castTr(sc_w, scwT, 384, 128);
    castTr(et_wn, etwnT, 384, 192);
    castTr(et_w1, etw1T, 448, 448); castTr(et_w2, etw2T, 448, 448); castTr(et_w3, etw3T, 448, 64);

    // ---- fuse + IPA projections ----
    maskf_kernel<<<dim3((NND + 255) / 256), dim3(256), 0, stream>>>(res_mask, maskf, NND);
    concat_fuse_kernel<<<dim3((unsigned)(((long)NND * 512 + 255) / 256)), dim3(256), 0, stream>>>(
        node_features, sidechain, Afuse16, NND);
    gemm(Afuse16, fuse_wT, fuse_b, nullptr, s16, NND, 384, 512, 0);
    gemm(s16, wqT, ipa_bq, Qf, nullptr, NND, 192, 384, 0);
    gemm(s16, wkT, ipa_bk, Kf, nullptr, NND, 192, 384, 0);
    gemm(s16, wvT, ipa_bv, Vf, nullptr, NND, 192, 384, 0);
    gemm(s16, wqpT, nullptr, qp, nullptr, NND, 144, 384, 0);
    gemm(s16, wkpT, nullptr, kp, nullptr, NND, 144, 384, 0);
    gemm(s16, wvpT, nullptr, vp, nullptr, NND, 288, 384, 0);
    to_global_kernel<<<dim3((NND * 48 + 255) / 256), dim3(256), 0, stream>>>(qp, rot, trans, 48, NND);
    to_global_kernel<<<dim3((NND * 48 + 255) / 256), dim3(256), 0, stream>>>(kp, rot, trans, 48, NND);
    to_global_kernel<<<dim3((NND * 96 + 255) / 256), dim3(256), 0, stream>>>(vp, rot, trans, 96, NND);
    cast16(edge_feat, zE16, (long)EED * 64);
    gemm(zE16, wbT, nullptr, zb, nullptr, EED, 16, 64, 0);   // N padded to 16

    // ---- IPA attention (segment softmax over edge targets) ----
    fillu(segmax, encNeg, (long)NND * 12);
    fillu((unsigned*)segden, 0u, (long)NND * 12);
    fillu((unsigned*)o_acc, 0u, (long)NND * 192);
    fillu((unsigned*)optg, 0u, (long)NND * 288);
    fillu((unsigned*)opair, 0u, (long)NND * 768);
    att_kernel<<<dim3(EED / 256), dim3(256), 0, stream>>>(Qf, Kf, qp, kp, zb, ipa_hw, maskf,
                                                          edge_index, att, segmax);
    exp_kernel<<<dim3((unsigned)(((long)EED * 12 + 255) / 256)), dim3(256), 0, stream>>>(
        edge_index, att, segmax, segden);
    accum_kernel<<<dim3((unsigned)(((long)EED * 12 + 255) / 256)), dim3(256), 0, stream>>>(
        edge_index, att, segden, Vf, vp, edge_feat, o_acc, optg, opair);
    finalize_cat_kernel<<<dim3((NND * 12 + 255) / 256), dim3(256), 0, stream>>>(
        o_acc, optg, opair, rot, trans, cat16);
    gemm(cat16, woT, ipa_bo, upd, nullptr, NND, 384, CATD, 0);
    add_rowmask_kernel<<<dim3((unsigned)(((long)NND * 384 + 255) / 256)), dim3(256), 0, stream>>>(
        node_features, upd, maskf, pre1, NND, 384);
    ln_kernel<<<dim3((NND + 7) / 8), dim3(256), 0, stream>>>(pre1, ln1_g, ln1_b, nullptr,
                                                             node1, node1_16, NND, 384);

    // ---- virtual-node attention ----
    cast16(vn_features, vn16, (long)BBD * 384);
    gemm(vn16, vnqT, nullptr, q2, nullptr, BBD, 384, 384, 0);
    gemm(node1_16, vnkT, nullptr, k2, nullptr, NND, 384, 384, 0);
    gemm(node1_16, vnvT, nullptr, v2, nullptr, NND, 384, 384, 0);
    fillu(bmax, encNeg, BBD * 12);
    fillu((unsigned*)bden, 0u, BBD * 12);
    fillu((unsigned*)vnout, 0u, (long)BBD * 384);
    vn_logits_kernel<<<dim3((NND * 12 + 255) / 256), dim3(256), 0, stream>>>(q2, k2, batch, maskf, lg2, bmax);
    vn_exp_kernel<<<dim3((NND * 12 + 255) / 256), dim3(256), 0, stream>>>(batch, lg2, bmax, bden);
    vn_accum_kernel<<<dim3((NND * 12 + 255) / 256), dim3(256), 0, stream>>>(batch, lg2, bden, v2, vnout);
    cast16(vnout, vnout16, (long)BBD * 384);
    gemm(vnout16, vnwoT, nullptr, vno_o, nullptr, BBD, 384, 384, 0);
    vn_add_kernel<<<dim3((BBD * 384 + 255) / 256), dim3(256), 0, stream>>>(
        vn_features, vno_o, o_vn, vnNew16, (long)BBD * 384);
    gemm(vnNew16, vnwbT, nullptr, vnb, nullptr, BBD, 384, 384, 0);
    gather_add_kernel<<<dim3((unsigned)(((long)NND * 384 + 255) / 256)), dim3(256), 0, stream>>>(
        node1, vnb, batch, maskf, node2, node2_16, NND, 384);

    // ---- node transition MLP + LN (masked) ----
    gemm(node2_16, ntw1T, nt_b1, nullptr, hA16, NND, 384, 384, 1);
    gemm(hA16, ntw2T, nt_b2, nullptr, hB16, NND, 384, 384, 1);
    gemm(hB16, ntw3T, nt_b3, ht, nullptr, NND, 384, 384, 0);
    add_kernel<<<dim3((unsigned)(((long)NND * 384 + 255) / 256)), dim3(256), 0, stream>>>(
        node2, ht, pre2, (long)NND * 384);
    ln_kernel<<<dim3((NND + 7) / 8), dim3(256), 0, stream>>>(pre2, nt_g, nt_bln, maskf,
                                                             o_node, nodeF16, NND, 384);

    // ---- backbone update + compose, sidechain ----
    make_nm16_kernel<<<dim3((unsigned)(((long)NND * 384 + 255) / 256)), dim3(256), 0, stream>>>(
        o_node, noising, nodeNM16, NND, 384);
    bb_compose_kernel<<<dim3((NND + 127) / 128), dim3(128), 0, stream>>>(
        o_node, noising, bb_w, bb_b, rot, trans, o_rot, o_tr, NND);
    gemm(nodeNM16, scwT, nullptr, scout, nullptr, NND, 128, 384, 0);
    sc_epilogue_kernel<<<dim3((unsigned)(((long)NND * 128 + 255) / 256)), dim3(256), 0, stream>>>(
        sidechain, scout, sc_b, noising, o_sc, NND, 128);

    // ---- fused edge transition (LDS-resident MLP, 128 edges/block, TDM z-staging) ----
    gemm(nodeF16, etwnT, et_bn2, nullptr, nb16, NND, 192, 384, 0);
    size_t edgeSm = (size_t)EB * HIDD * 2 * 2 + (size_t)EB * CZD * 2;   // 245760 B
    edge_mlp_kernel<<<dim3(EED / EB), dim3(256), edgeSm, stream>>>(
        zE16, nb16, edge_index, etw1T, et_b1, etw2T, et_b2, etw3T, et_b3,
        et_g, et_bln, o_edge);
}
